// NODEAttention_4466765988284
// MI455X (gfx1250) — compile-verified, matrix operand order checked by reading, not run on hardware
//
#include <hip/hip_runtime.h>
#include <hip/hip_bf16.h>

// ============================================================================
// NODE attention block for MI455X (gfx1250).
// All matmuls via v_wmma_f32_16x16x32_bf16 (fp32 accumulate).
// ============================================================================

typedef __bf16 bf16_t;
typedef __attribute__((ext_vector_type(16))) __bf16 v16bf;
typedef __attribute__((ext_vector_type(8)))  float  v8f;

constexpr int CB  = 16;           // batch
constexpr int CNL = 512;          // node tokens
constexpr int CEL = 1024;         // edge tokens
constexpr int CD  = 256;          // model dim
constexpr int CH  = 8;            // heads
constexpr int CHD = 32;           // head dim
constexpr int CKL = CNL + CEL;    // 1536 keys
constexpr int MWORDS = CKL / 32;  // 48 mask words per query row
constexpr float CEPS = 1e-5f;
#define QSCALE 0.17677669529663687f  /* 32^-0.5 */

// ---------------- bf16 helpers (round-to-nearest-even) ----------------------
__device__ inline bf16_t to_bf16(float x) {
  union { unsigned short u; bf16_t b; } cv;
  unsigned int t = __float_as_uint(x);
  cv.u = (unsigned short)((t + 0x7FFFu + ((t >> 16) & 1u)) >> 16);
  return cv.b;
}
__device__ inline float bf16_to_f32(bf16_t b) {
  union { bf16_t b; unsigned short u; } cv; cv.b = b;
  return __uint_as_float(((unsigned int)cv.u) << 16);
}

// ---------------- WMMA fragment loaders -------------------------------------
// A-fragment 16x32 bf16 from row-major [16 x ld] storage.
// ISA layout: lanes 0-15 hold M=lane, K in {kb..kb+7, kb+16..kb+23}, kb=0;
// lanes 16-31 same M with kb=8.
__device__ inline v16bf load_frag_rm(const bf16_t* __restrict__ s, int ld, int lane) {
  int r  = lane & 15;
  int kb = (lane & 16) ? 8 : 0;
  const bf16_t* p = s + r * ld + kb;
  v16bf f;
#pragma unroll
  for (int i = 0; i < 8; ++i) f[i] = p[i];
#pragma unroll
  for (int i = 0; i < 8; ++i) f[8 + i] = p[16 + i];
  return f;
}
// B-fragment 32x16 bf16 from row-major [32 x ld] (K-major) storage; lane = N.
__device__ inline v16bf load_frag_kn(const bf16_t* __restrict__ s, int ld, int lane) {
  int n  = lane & 15;
  int kb = (lane & 16) ? 8 : 0;
  v16bf f;
#pragma unroll
  for (int i = 0; i < 8; ++i) f[i]     = s[(kb + i) * ld + n];
#pragma unroll
  for (int i = 0; i < 8; ++i) f[8 + i] = s[(kb + 16 + i) * ld + n];
  return f;
}
__device__ inline v8f wmma_bf16(v16bf a, v16bf b, v8f c) {
  return __builtin_amdgcn_wmma_f32_16x16x32_bf16(false, a, false, b, (short)0, c,
                                                 false, false);
}

// ---------------- elementwise conversion / packing ---------------------------
__global__ void f2bf_kernel(const float* __restrict__ src, bf16_t* __restrict__ dst, int n) {
  int i = blockIdx.x * blockDim.x + threadIdx.x;
  if (i < n) dst[i] = to_bf16(src[i]);
}

__global__ void pack_mask_kernel(const unsigned char* __restrict__ m,
                                 unsigned int* __restrict__ bits, int nwords) {
  int w = blockIdx.x * blockDim.x + threadIdx.x;
  if (w >= nwords) return;
  const unsigned char* p = m + (size_t)w * 32;
  unsigned int r = 0;
#pragma unroll
  for (int i = 0; i < 32; ++i) r |= (p[i] ? 1u : 0u) << i;
  bits[w] = r;
}

// ---------------- generic tiled WMMA GEMM ------------------------------------
// C[M,N] = A[M,K](bf16) * B(bf16) + bias, optional ReLU, fp32 or bf16 output.
// B_IS_NK=false: B stored [K,N] row-major (x @ W).  true: [N,K] (x @ W^T).
template <bool B_IS_NK, bool RELU, bool OUT_BF16>
__global__ __launch_bounds__(256) void gemm_bf16_kernel(
    const bf16_t* __restrict__ A, const bf16_t* __restrict__ Bm,
    const float* __restrict__ bias, void* __restrict__ Cout,
    int M, int N, int K) {
  (void)M;
  __shared__ __align__(16) bf16_t As[128 * 32];
  __shared__ __align__(16) bf16_t Bs[128 * 32];
  int tid = threadIdx.x, lane = tid & 31, wave = tid >> 5;
  int m0 = blockIdx.x * 128, n0 = blockIdx.y * 128;
  int wm = (wave & 3) * 32;    // 4 waves along M, 32 rows each
  int wn = (wave >> 2) * 64;   // 2 waves along N, 64 cols each
  v8f acc[2][4] = {};

  for (int k0 = 0; k0 < K; k0 += 32) {
    __syncthreads();
    {  // stage A tile [128 x 32]
      int r = tid >> 1, hf = (tid & 1) * 16;
      const uint4* src = (const uint4*)(A + (size_t)(m0 + r) * K + k0 + hf);
      uint4* dst = (uint4*)(As + r * 32 + hf);
      dst[0] = src[0]; dst[1] = src[1];
    }
    if (B_IS_NK) {  // B tile is [128 n-rows x 32 k-cols]
      int r = tid >> 1, hf = (tid & 1) * 16;
      const uint4* src = (const uint4*)(Bm + (size_t)(n0 + r) * K + k0 + hf);
      uint4* dst = (uint4*)(Bs + r * 32 + hf);
      dst[0] = src[0]; dst[1] = src[1];
    } else {        // B tile is [32 k-rows x 128 n-cols]
      int kk = tid >> 3, seg = (tid & 7) * 16;
      const uint4* src = (const uint4*)(Bm + (size_t)(k0 + kk) * N + n0 + seg);
      uint4* dst = (uint4*)(Bs + kk * 128 + seg);
      dst[0] = src[0]; dst[1] = src[1];
    }
    __syncthreads();

    v16bf afr[2], bfr[4];
    afr[0] = load_frag_rm(As + (wm + 0)  * 32, 32, lane);
    afr[1] = load_frag_rm(As + (wm + 16) * 32, 32, lane);
#pragma unroll
    for (int j = 0; j < 4; ++j)
      bfr[j] = B_IS_NK ? load_frag_rm(Bs + (wn + j * 16) * 32, 32, lane)
                       : load_frag_kn(Bs + wn + j * 16, 128, lane);
#pragma unroll
    for (int i = 0; i < 2; ++i)
#pragma unroll
      for (int j = 0; j < 4; ++j)
        acc[i][j] = wmma_bf16(afr[i], bfr[j], acc[i][j]);
  }

  // epilogue: C layout = VGPR e -> row e (+8 for upper half-wave), col = lane&15
  int mrow = (lane & 16) ? 8 : 0;
  int ncol = lane & 15;
#pragma unroll
  for (int i = 0; i < 2; ++i)
#pragma unroll
    for (int j = 0; j < 4; ++j) {
      int nbase = n0 + wn + j * 16 + ncol;
      float bv = bias[nbase];
#pragma unroll
      for (int e = 0; e < 8; ++e) {
        int row = m0 + wm + i * 16 + mrow + e;
        float v = acc[i][j][e] + bv;
        if (RELU) v = fmaxf(v, 0.0f);
        if (OUT_BF16) ((bf16_t*)Cout)[(size_t)row * N + nbase] = to_bf16(v);
        else          ((float*) Cout)[(size_t)row * N + nbase] = v;
      }
    }
}

// ---------------- head repacking ---------------------------------------------
// qkv [B*NL, 768] -> Qh [B*H][NL][32] (pre-scaled), Kh/Vh [B*H][KL][32] rows 0..NL
__global__ void repack_node_kernel(const bf16_t* __restrict__ qkv,
                                   bf16_t* __restrict__ Qh,
                                   bf16_t* __restrict__ Kh,
                                   bf16_t* __restrict__ Vh) {
  int idx = blockIdx.x * 256 + threadIdx.x;       // over B*NL*D = 2^21
  int d  = idx & (CD - 1);
  int n  = (idx >> 8) & (CNL - 1);
  int b  = idx >> 17;
  int h  = d >> 5, dh = d & 31;
  size_t row = (size_t)b * CNL + n;
  size_t ho  = ((size_t)b * CH + h);
  Qh[(ho * CNL + n) * CHD + dh] = to_bf16(bf16_to_f32(qkv[row * 768 + d]) * QSCALE);
  Kh[(ho * CKL + n) * CHD + dh] = qkv[row * 768 + CD + d];
  Vh[(ho * CKL + n) * CHD + dh] = qkv[row * 768 + 2 * CD + d];
}
// kve [B*EL, 512] -> Kh/Vh rows NL..NL+EL
__global__ void repack_edge_kernel(const bf16_t* __restrict__ kve,
                                   bf16_t* __restrict__ Kh,
                                   bf16_t* __restrict__ Vh) {
  int idx = blockIdx.x * 256 + threadIdx.x;       // over B*EL*D = 2^22
  int d  = idx & (CD - 1);
  int e  = (idx >> 8) & (CEL - 1);
  int b  = idx >> 18;
  int h  = d >> 5, dh = d & 31;
  size_t row = (size_t)b * CEL + e;
  size_t ho  = ((size_t)b * CH + h);
  Kh[(ho * CKL + CNL + e) * CHD + dh] = kve[row * 512 + d];
  Vh[(ho * CKL + CNL + e) * CHD + dh] = kve[row * 512 + CD + d];
}

// ---------------- flash-style attention --------------------------------------
// grid.x = B*H, grid.y = NL/128. 8 waves/block, one 16-query tile per wave.
__global__ __launch_bounds__(256) void attn_kernel(
    const bf16_t* __restrict__ Qh, const bf16_t* __restrict__ Kh,
    const bf16_t* __restrict__ Vh, const unsigned int* __restrict__ mbits,
    float* __restrict__ ctx) {
  __shared__ __align__(16) bf16_t Kt[32 * 32];
  __shared__ __align__(16) bf16_t Vt[32 * 32];
  __shared__ __align__(16) bf16_t Ps[8][16 * 32];
  int tid = threadIdx.x, lane = tid & 31, wave = tid >> 5;
  int bh = blockIdx.x;
  int b = bh >> 3, h = bh & 7;
  int q0 = blockIdx.y * 128 + wave * 16;

  v16bf qf = load_frag_rm(Qh + ((size_t)bh * CNL + q0) * CHD, CHD, lane);
  v8f o0 = {}, o1 = {};
  float mrow[8], lrow[8];
#pragma unroll
  for (int j = 0; j < 8; ++j) { mrow[j] = -__builtin_inff(); lrow[j] = 0.0f; }
  int mr = (lane & 16) ? 8 : 0;
  int nc = lane & 15;
  const unsigned int* mbase = mbits + (size_t)b * CNL * MWORDS;

  for (int kb = 0; kb < CKL; kb += 32) {
    {  // cooperatively stage K/V chunk [32 keys x 32 dims]
      int r = tid >> 3, seg = (tid & 7) * 4;
      size_t src = ((size_t)bh * CKL + kb + r) * CHD + seg;
      *(uint2*)(Kt + r * 32 + seg) = *(const uint2*)(Kh + src);
      *(uint2*)(Vt + r * 32 + seg) = *(const uint2*)(Vh + src);
    }
    __syncthreads();

    // scores: S = q(16x32) * K^T(32x16) for two 16-key halves
    v8f z = {};
    v8f s0 = wmma_bf16(qf, load_frag_rm(Kt,           32, lane), z);
    v8f s1 = wmma_bf16(qf, load_frag_rm(Kt + 16 * 32, 32, lane), z);

    // online softmax; each row lives in one half-wave (16 lanes), VGPR j
#pragma unroll
    for (int j = 0; j < 8; ++j) {
      int q = q0 + j + mr;
      unsigned int w = mbase[(size_t)q * MWORDS + (kb >> 5)];
      float v0 = ((w >> nc)        & 1u) ? -__builtin_inff() : s0[j];
      float v1 = ((w >> (nc + 16)) & 1u) ? -__builtin_inff() : s1[j];
      float mx = fmaxf(v0, v1);
      mx = fmaxf(mx, __shfl_xor(mx, 1));
      mx = fmaxf(mx, __shfl_xor(mx, 2));
      mx = fmaxf(mx, __shfl_xor(mx, 4));
      mx = fmaxf(mx, __shfl_xor(mx, 8));
      float mnew  = fmaxf(mrow[j], mx);
      float alpha = __expf(mrow[j] - mnew);
      float p0 = __expf(v0 - mnew);
      float p1 = __expf(v1 - mnew);
      float ps = p0 + p1;
      ps += __shfl_xor(ps, 1);
      ps += __shfl_xor(ps, 2);
      ps += __shfl_xor(ps, 4);
      ps += __shfl_xor(ps, 8);
      lrow[j] = lrow[j] * alpha + ps;
      mrow[j] = mnew;
      o0[j] = o0[j] * alpha;
      o1[j] = o1[j] * alpha;
      Ps[wave][(j + mr) * 32 + nc]      = to_bf16(p0);
      Ps[wave][(j + mr) * 32 + nc + 16] = to_bf16(p1);
    }
    asm volatile("s_wait_dscnt 0x0" ::: "memory");  // P LDS write -> A-frag read

    // ctx: O += P(16x32keys) * V(32keys x 32dims), two 16-dim halves
    v16bf pf = load_frag_rm(Ps[wave], 32, lane);
    o0 = wmma_bf16(pf, load_frag_kn(Vt,      32, lane), o0);
    o1 = wmma_bf16(pf, load_frag_kn(Vt + 16, 32, lane), o1);
    __syncthreads();
  }

#pragma unroll
  for (int j = 0; j < 8; ++j) {
    int q = q0 + j + mr;
    float inv = 1.0f / lrow[j];
    float* dst = ctx + ((size_t)b * CNL + q) * CD + h * CHD;
    dst[nc]      = o0[j] * inv;
    dst[nc + 16] = o1[j] * inv;
  }
}

// ---------------- residual + LayerNorm ---------------------------------------
__global__ __launch_bounds__(256) void ln_kernel(
    const float* __restrict__ x, const float* __restrict__ y,
    const float* __restrict__ g, const float* __restrict__ beta,
    float* __restrict__ out, bf16_t* __restrict__ out_bf) {
  __shared__ float red[256];
  int row = blockIdx.x, t = threadIdx.x;
  float v = x[(size_t)row * CD + t] + y[(size_t)row * CD + t];
  red[t] = v;
  __syncthreads();
  for (int s = 128; s > 0; s >>= 1) { if (t < s) red[t] += red[t + s]; __syncthreads(); }
  float mu = red[0] * (1.0f / CD);
  __syncthreads();
  float c = v - mu;
  red[t] = c * c;
  __syncthreads();
  for (int s = 128; s > 0; s >>= 1) { if (t < s) red[t] += red[t + s]; __syncthreads(); }
  float var = red[0] * (1.0f / CD);
  float o = c * rsqrtf(var + CEPS) * g[t] + beta[t];
  out[(size_t)row * CD + t] = o;
  if (out_bf) out_bf[(size_t)row * CD + t] = to_bf16(o);
}

// ============================================================================
extern "C" void kernel_launch(void* const* d_in, const int* in_sizes, int n_in,
                              void* d_out, int out_size, void* d_ws, size_t ws_size,
                              hipStream_t stream) {
  (void)in_sizes; (void)n_in; (void)out_size; (void)ws_size;
  const float*         node_x  = (const float*)d_in[0];
  const float*         edge_x  = (const float*)d_in[1];
  const unsigned char* mask    = (const unsigned char*)d_in[2];
  const float* W1 = (const float*)d_in[3];  const float* b1 = (const float*)d_in[4];
  const float* W2 = (const float*)d_in[5];  const float* b2 = (const float*)d_in[6];
  const float* lin1_w = (const float*)d_in[7];  const float* lin1_b = (const float*)d_in[8];
  const float* lin2_w = (const float*)d_in[9];  const float* lin2_b = (const float*)d_in[10];
  const float* ln1_g = (const float*)d_in[11];  const float* ln1_bta = (const float*)d_in[12];
  const float* ln2_g = (const float*)d_in[13];  const float* ln2_bta = (const float*)d_in[14];

  char* ws = (char*)d_ws;
  size_t off = 0;
  auto take = [&](size_t bytes) -> void* {
    void* p = ws + off;
    off += (bytes + 255) & ~(size_t)255;
    return p;
  };
  bf16_t* Xn   = (bf16_t*)take((size_t)CB * CNL * CD * 2);          // 4 MB
  bf16_t* Xe   = (bf16_t*)take((size_t)CB * CEL * CD * 2);          // 8 MB
  bf16_t* W1b  = (bf16_t*)take((size_t)CD * 3 * CD * 2);
  bf16_t* W2b  = (bf16_t*)take((size_t)CD * 2 * CD * 2);
  bf16_t* L1b  = (bf16_t*)take((size_t)CD * CD * 2);
  bf16_t* L2b  = (bf16_t*)take((size_t)CD * CD * 2);
  bf16_t* qkv  = (bf16_t*)take((size_t)CB * CNL * 3 * CD * 2);      // 12.6 MB
  bf16_t* kve  = (bf16_t*)take((size_t)CB * CEL * 2 * CD * 2);      // 16.8 MB
  bf16_t* Qh   = (bf16_t*)take((size_t)CB * CH * CNL * CHD * 2);    // 4 MB
  bf16_t* Kh   = (bf16_t*)take((size_t)CB * CH * CKL * CHD * 2);    // 12.6 MB
  bf16_t* Vh   = (bf16_t*)take((size_t)CB * CH * CKL * CHD * 2);    // 12.6 MB
  unsigned int* mbits = (unsigned int*)take((size_t)CB * CNL * MWORDS * 4);
  // aliased buffers (dead source regions by the time they are written):
  float*  ctx = (float*)kve;   // ctx fp32 8.4 MB  <= kve 16.8 MB (dead after repack)
  float*  x1  = (float*)Xe;    // x1  fp32 8.4 MB  <= Xe  8.4 MB  (dead after kve gemm)
  bf16_t* x1b = Xn;            // 4 MB            <= Xn           (dead after qkv gemm)
  bf16_t* ffh = Qh;            // 4 MB            <= Qh           (dead after attn)
  float*  ff  = (float*)qkv;   // 8.4 MB          <= qkv 12.6 MB  (dead after repack)

  // 1) convert to bf16 + pack mask
  f2bf_kernel<<<(CB * CNL * CD + 255) / 256, 256, 0, stream>>>(node_x, Xn, CB * CNL * CD);
  f2bf_kernel<<<(CB * CEL * CD + 255) / 256, 256, 0, stream>>>(edge_x, Xe, CB * CEL * CD);
  f2bf_kernel<<<(CD * 3 * CD + 255) / 256, 256, 0, stream>>>(W1, W1b, CD * 3 * CD);
  f2bf_kernel<<<(CD * 2 * CD + 255) / 256, 256, 0, stream>>>(W2, W2b, CD * 2 * CD);
  f2bf_kernel<<<(CD * CD + 255) / 256, 256, 0, stream>>>(lin1_w, L1b, CD * CD);
  f2bf_kernel<<<(CD * CD + 255) / 256, 256, 0, stream>>>(lin2_w, L2b, CD * CD);
  pack_mask_kernel<<<(CB * CNL * MWORDS + 255) / 256, 256, 0, stream>>>(
      mask, mbits, CB * CNL * MWORDS);

  // 2) projections: qkv = Xn@W1+b1  |  kve = Xe@W2+b2   (B stored [K,N])
  gemm_bf16_kernel<false, false, true><<<dim3(64, 6), 256, 0, stream>>>(
      Xn, W1b, b1, qkv, CB * CNL, 3 * CD, CD);
  gemm_bf16_kernel<false, false, true><<<dim3(128, 4), 256, 0, stream>>>(
      Xe, W2b, b2, kve, CB * CEL, 2 * CD, CD);

  // 3) head repack (+ fold 1/sqrt(HD) into Q)
  repack_node_kernel<<<CB * CNL * CD / 256, 256, 0, stream>>>(qkv, Qh, Kh, Vh);
  repack_edge_kernel<<<CB * CEL * CD / 256, 256, 0, stream>>>(kve, Kh, Vh);

  // 4) attention
  attn_kernel<<<dim3(CB * CH, CNL / 128), 256, 0, stream>>>(Qh, Kh, Vh, mbits, ctx);

  // 5) residual + LN1 (fp32 + bf16 copy for FFN)
  ln_kernel<<<CB * CNL, 256, 0, stream>>>(node_x, ctx, ln1_g, ln1_bta, x1, x1b);

  // 6) FFN: relu(x1 @ lin1_w^T + b) @ lin2_w^T + b   (B stored [N,K])
  gemm_bf16_kernel<true, true, true><<<dim3(64, 2), 256, 0, stream>>>(
      x1b, L1b, lin1_b, ffh, CB * CNL, CD, CD);
  gemm_bf16_kernel<true, false, false><<<dim3(64, 2), 256, 0, stream>>>(
      ffh, L2b, lin2_b, ff, CB * CNL, CD, CD);

  // 7) residual + LN2 -> output
  ln_kernel<<<CB * CNL, 256, 0, stream>>>(x1, ff, ln2_g, ln2_bta, (float*)d_out, nullptr);
}
